// ScaledDotProductAttention_36386962932539
// MI455X (gfx1250) — matrix operand order repaired and checked by reading, not compiled, not run on hardware
//
#include <hip/hip_runtime.h>
#include <hip/hip_bf16.h>

// ---------------------------------------------------------------------------
// Attention forward for B=4, H=16, S=1024, D=128 (fp32 in, fp32 out),
// returning both out[B,H,S,D] and attn[B,H,S,S] (concatenated in d_out).
// attn = clamp(where(mask==0, -1e9, (Q/T)K^T), 0, 15);  out = attn @ V
// f16 WMMA (v_wmma_f32_16x16x32_f16) with f32 accumulation.
// K tile staged once per WG in LDS (f16); mask tile fetched via TDM
// (tensor_load_to_lds, 6-arg toolchain form); V^T staged in LDS for GEMM2.
// ---------------------------------------------------------------------------

typedef _Float16 v16h __attribute__((ext_vector_type(16)));
typedef _Float16 v8h  __attribute__((ext_vector_type(8)));
typedef float    v8f  __attribute__((ext_vector_type(8)));
typedef unsigned int u32x4 __attribute__((ext_vector_type(4)));
typedef int          i32x4 __attribute__((ext_vector_type(4)));
typedef int          i32x8 __attribute__((ext_vector_type(8)));

#define SEQ    1024
#define DIM    128
#define NHEAD  16
#define NBATCH 4
#define TEMP_INV 0.08838834764831845f   /* 1/sqrt(128) */
#define CAP 15.0f

// LDS strides in halves; multiples of 8 halves (16B) for aligned b128 access
#define VT_STRIDE 40    // V^T tile rows (80B)
#define KT_STRIDE 136   // K tile rows (272B)
#define P_STRIDE  40    // P tile rows (80B)

#if __has_builtin(__builtin_amdgcn_tensor_load_to_lds) && __has_builtin(__builtin_amdgcn_s_wait_tensorcnt)
#define USE_TDM_MASK 1
#else
#define USE_TDM_MASK 0
#endif

__global__ __launch_bounds__(128)
void ScaledDotProductAttention_36386962932539_kernel(
    const float* __restrict__ q, const float* __restrict__ k,
    const float* __restrict__ v, const int* __restrict__ mask,
    float* __restrict__ outO, float* __restrict__ outA)
{
    __shared__ __align__(16) _Float16 VT[DIM * VT_STRIDE];   // V^T tile [d][s]
    __shared__ __align__(16) _Float16 KT[32 * KT_STRIDE];    // K tile  [s][d]
    __shared__ __align__(16) _Float16 PL[4][16 * P_STRIDE];  // per-wave P [q][n]
    __shared__ __align__(16) int      LM[64 * 32];           // mask tile [q][n]

    const int tid  = threadIdx.x;
    const int wave = tid >> 5;
    const int lane = tid & 31;
    const int l16  = lane & 15;
    const int hi   = (lane >> 4) & 1;   // 0: lanes 0-15, 1: lanes 16-31

    const int bh = blockIdx.x >> 4;           // 0..63  (b*H + h)
    const int qt = blockIdx.x & 15;           // q tile of 64 rows
    const int b  = bh >> 4;                   // batch index (H=16)
    const int qWG   = qt * 64;                // workgroup's first Q row
    const int qBase = qWG + wave * 16;        // this wave's 16 Q rows

    const size_t matBase = (size_t)bh * SEQ * DIM;
    const float* kbase_ptr = k + matBase;
    const float* vbase_ptr = v + matBase;
    float*       oA   = outA + (size_t)bh * SEQ * SEQ;
    const int*   mrow = mask + (size_t)b  * SEQ * SEQ;

    // ---- Q fragments: A-layout 16x32 f16, pre-scaled by 1/T --------------
    // A 16x32 in this code's contiguous k order (B matches): lanes 0-15 row M=l16 hold K {0..15}; lanes 16-31
    // hold K {16..31}  -> one contiguous 16-float chunk per lane.
    v16h qa[4];
    {
        const float* qrow = q + matBase + (size_t)(qBase + l16) * DIM;
        const int off = hi ? 16 : 0;
        #pragma unroll
        for (int dc = 0; dc < 4; ++dc) {
            const float* p0 = qrow + dc * 32 + off;
            v16h a;
            #pragma unroll
            for (int e = 0; e < 8; ++e) a[e]     = (_Float16)(p0[e]      * TEMP_INV);
            #pragma unroll
            for (int e = 0; e < 8; ++e) a[e + 8] = (_Float16)(p0[e + 8] * TEMP_INV);
            qa[dc] = a;
        }
    }

    // ---- O accumulators: 16x128 f32 as 8 WMMA C/D fragments --------------
    const v8f vzero = {};
    v8f oacc[8];
    #pragma unroll
    for (int i = 0; i < 8; ++i) oacc[i] = vzero;

    _Float16* myP = &PL[wave][0];
    const int* lm = &LM[wave * 16 * 32];

    for (int kb = 0; kb < SEQ; kb += 32) {
        if (kb + 32 < SEQ) {  // uniform branch; warm next tiles into caches
            __builtin_prefetch(kbase_ptr + (size_t)(kb + 32) * DIM + (tid << 2), 0, 0);
            __builtin_prefetch(vbase_ptr + (size_t)(kb + 32) * DIM + (tid << 2), 0, 0);
        }

        __syncthreads();   // protect previous iteration's LDS reads

        // ---- stage K tile (f16) into LDS: KT[s][d], 32 floats/thread -----
        {
            const int  row = tid >> 2;          // 0..31
            const int  ch  = (tid & 3) * 32;    // 0,32,64,96
            const float* kp = kbase_ptr + (size_t)(kb + row) * DIM + ch;
            _Float16* dst = &KT[row * KT_STRIDE + ch];
            #pragma unroll
            for (int c = 0; c < 4; ++c) {
                v8h hch;
                #pragma unroll
                for (int e = 0; e < 8; ++e) hch[e] = (_Float16)kp[c * 8 + e];
                *(v8h*)(dst + c * 8) = hch;
            }
        }

        // ---- stage V^T (f16) into LDS: VT[d][s], coalesced reads ---------
        {
            const float* vt = vbase_ptr + (size_t)kb * DIM + tid;   // column tid
            #pragma unroll 8
            for (int s2 = 0; s2 < 32; ++s2)
                VT[tid * VT_STRIDE + s2] = (_Float16)vt[(size_t)s2 * DIM];
        }

#if USE_TDM_MASK
        // ---- mask tile [qWG..qWG+63] x [kb..kb+31] via Tensor Data Mover -
        if (wave == 0) {
            const unsigned long long ga =
                (unsigned long long)(const void*)(mrow + (size_t)qWG * SEQ + kb);
            const unsigned lds_a = (unsigned)(unsigned long long)(const void*)&LM[0];
            // D# group 0: count=1 | lds_addr | global_addr | type=2
            u32x4 g0;
            g0[0] = 1u;                                   // count=1, user desc
            g0[1] = lds_a;                                // LDS byte address
            g0[2] = (unsigned)ga;                         // global addr [31:0]
            g0[3] = (unsigned)((ga >> 32) & 0x01FFFFFFu) | (2u << 30); // type=2
            // D# group 1: data_size=4B, dims large, tile 32x64, stride0=SEQ
            i32x8 g1;
            g1[0] = (int)(2u << 16);          // data_size=2 (4 bytes)
            g1[1] = 0;                        // tensor_dim0[15:0]=0 (dim0=1<<24)
            g1[2] = (int)0x00000100u;         // dim0[31:16]=0x100, dim1[15:0]=0
            g1[3] = (int)((32u << 16) | 0x100u); // dim1[31:16]=0x100, tile_dim0=32
            g1[4] = 64;                       // tile_dim1=64, tile_dim2=0
            g1[5] = SEQ;                      // tensor_dim0_stride = 1024 elems
            g1[6] = 0;
            g1[7] = 0;
            const i32x4 gz4 = {0, 0, 0, 0};          // groups 2/3 unused (2D)
            const i32x8 gz8 = {0, 0, 0, 0, 0, 0, 0, 0};
            __builtin_amdgcn_tensor_load_to_lds(g0, g1, gz4, gz4, gz8, 0);
            __builtin_amdgcn_s_wait_tensorcnt(0);
        }
#else
        // fallback: cooperative global fill of LM (16 ints/thread)
        {
            const int row = tid >> 1;               // 0..63
            const int c0  = (tid & 1) * 16;
            const int* mp = mrow + (size_t)(qWG + row) * SEQ + kb + c0;
            #pragma unroll
            for (int e = 0; e < 16; ++e) LM[row * 32 + c0 + e] = mp[e];
        }
#endif
        __syncthreads();

        // ---- GEMM1: S16x32 = Q(16x128) . K^T  via 8 WMMAs ----------------
        // B-layout 32x16: lane l -> column N=l16, K-dim(d) = dlo..dlo+15.
        v8f acc0 = vzero, acc1 = vzero;
        const int dlo = hi ? 16 : 0;
        #pragma unroll
        for (int dc = 0; dc < 4; ++dc) {
            const _Float16* kr0 = &KT[l16 * KT_STRIDE + dc * 32 + dlo];
            const _Float16* kr1 = kr0 + 16 * KT_STRIDE;
            const v16h b0 = __builtin_shufflevector(*(const v8h*)kr0, *(const v8h*)(kr0 + 8),
                                                    0,1,2,3,4,5,6,7,8,9,10,11,12,13,14,15);
            const v16h b1 = __builtin_shufflevector(*(const v8h*)kr1, *(const v8h*)(kr1 + 8),
                                                    0,1,2,3,4,5,6,7,8,9,10,11,12,13,14,15);
            acc0 = __builtin_amdgcn_wmma_f32_16x16x32_f16(false, qa[dc], false, b0,
                                                          (short)0, acc0, false, false);
            acc1 = __builtin_amdgcn_wmma_f32_16x16x32_f16(false, qa[dc], false, b1,
                                                          (short)0, acc1, false, false);
        }

        // ---- epilogue: mask, clamp; stream attn out; stash P in LDS ------
        // C/D layout: lane l reg r -> (q = r + 8*hi, n = l16)
        #pragma unroll
        for (int r = 0; r < 8; ++r) {
            const int qq   = (hi << 3) + r;          // local q row 0..15
            const int grow = qBase + qq;             // global q row
            const int kk0  = kb + l16;
            const int kk1  = kk0 + 16;
            float s0v = acc0[r], s1v = acc1[r];
            s0v = (lm[qq * 32 + l16]      == 0) ? -1e9f : s0v;
            s1v = (lm[qq * 32 + l16 + 16] == 0) ? -1e9f : s1v;
            s0v = fminf(fmaxf(s0v, 0.0f), CAP);
            s1v = fminf(fmaxf(s1v, 0.0f), CAP);
            __builtin_nontemporal_store(s0v, &oA[(size_t)grow * SEQ + kk0]);
            __builtin_nontemporal_store(s1v, &oA[(size_t)grow * SEQ + kk1]);
            myP[qq * P_STRIDE + l16]      = (_Float16)s0v;
            myP[qq * P_STRIDE + l16 + 16] = (_Float16)s1v;
        }
        // per-wave LDS RAW: DS ops are in-order per wave; wait for stores
        asm volatile("s_wait_dscnt 0" ::: "memory");

        // ---- re-layout P: D-fragment -> A-fragment (16x32 f16) -----------
        v16h pfrag;
        {
            const int c0 = hi ? 16 : 0;
            const v8h lo = *(const v8h*)&myP[l16 * P_STRIDE + c0];
            const v8h hh = *(const v8h*)&myP[l16 * P_STRIDE + c0 + 8];
            pfrag = __builtin_shufflevector(lo, hh, 0,1,2,3,4,5,6,7,
                                                    8,9,10,11,12,13,14,15);
        }

        // ---- GEMM2: O(16x128) += P(16x32) . V(32x128) via 8 WMMAs --------
        const int soff = hi ? 16 : 0;
        #pragma unroll
        for (int nc = 0; nc < 8; ++nc) {
            const _Float16* vp = &VT[(nc * 16 + l16) * VT_STRIDE + soff];
            const v16h bv = __builtin_shufflevector(*(const v8h*)vp, *(const v8h*)(vp + 8),
                                                    0,1,2,3,4,5,6,7,8,9,10,11,12,13,14,15);
            oacc[nc] = __builtin_amdgcn_wmma_f32_16x16x32_f16(false, pfrag, false, bv,
                                                              (short)0, oacc[nc], false, false);
        }
    }

    // ---- write O fragments: out[b,h,q,d] ---------------------------------
    float* oO = outO + matBase;
    #pragma unroll
    for (int nc = 0; nc < 8; ++nc) {
        #pragma unroll
        for (int r = 0; r < 8; ++r) {
            const int qq = qBase + (hi << 3) + r;
            const int dd = nc * 16 + l16;
            __builtin_nontemporal_store(oacc[nc][r], &oO[(size_t)qq * DIM + dd]);
        }
    }
}

extern "C" void kernel_launch(void* const* d_in, const int* in_sizes, int n_in,
                              void* d_out, int out_size, void* d_ws, size_t ws_size,
                              hipStream_t stream) {
    const float* q    = (const float*)d_in[0];
    const float* k    = (const float*)d_in[1];
    const float* v    = (const float*)d_in[2];
    const int*   mask = (const int*)d_in[3];

    float* outO = (float*)d_out;                                   // [B,H,S,D]
    float* outA = outO + (size_t)NBATCH * NHEAD * SEQ * DIM;       // [B,H,S,S]

    const int blocks = NBATCH * NHEAD * (SEQ / 64);   // 1024 WGs
    ScaledDotProductAttention_36386962932539_kernel<<<blocks, 128, 0, stream>>>(
        q, k, v, mask, outO, outA);
}
